// TransformerDecoderLayer_35536559407330
// MI455X (gfx1250) — compile-verified, matrix operand order checked by reading, not run on hardware
//
#include <hip/hip_runtime.h>
#include <math.h>

// ---------------------------------------------------------------------------
// Types for CDNA5 WMMA (gfx1250, wave32)
// ---------------------------------------------------------------------------
typedef __bf16 bf16_t;
typedef __attribute__((ext_vector_type(16))) __bf16 v16bf;
typedef __attribute__((ext_vector_type(8)))  __bf16 bf16x8;
typedef __attribute__((ext_vector_type(4)))  __bf16 bf16x4;
typedef __attribute__((ext_vector_type(8)))  float  v8f;

#define WMMA_BF16(a, b, c) \
  __builtin_amdgcn_wmma_f32_16x16x32_bf16(false, (a), false, (b), (short)0, (c), false, false)

// ---------------------------------------------------------------------------
// Problem constants (from reference)
// ---------------------------------------------------------------------------
#define DMODEL 512
#define NHEAD  8
#define DHEAD  64
#define NEXP   32
#define TOPK   4
#define NSH    8
#define RRANK  64
#define GDIM   64
#define GIN_   2048
#define KIN_   1024
#define VIN_   512
#define BATCH  4
#define NQTOK  1024
#define NTOK   4096            // BATCH * NQTOK
#define OUT_ELEMS 2097152      // 4*1024*512
#define AUX_OFF   2097152
#define OG_OFF    2097153

#define LDK 40                 // padded LDS row (elements); 80 B = multiple of 16

// ---------------------------------------------------------------------------
// Device helpers
// ---------------------------------------------------------------------------
__device__ __forceinline__ float gelu_f(float x) {
  float x3 = x * x * x;
  return 0.5f * x * (1.f + tanhf(0.7978845608028654f * (x + 0.044715f * x3)));
}
__device__ __forceinline__ float sigmoid_f(float x) { return 1.f / (1.f + expf(-x)); }

// Fragment load from an LDS tile laid out row-major along K with leading dim ld.
// Works for both A (row = M) and B (row = N) operands of
// v_wmma_f32_16x16x32_bf16 (ISA 7.12.2): lane (r=lane&15, hi=lane>>4) needs
// K runs [k0+hi*8, +8) and [k0+16+hi*8, +8) of its row -> two ds_load_b128.
__device__ __forceinline__ v16bf frag_row(const bf16_t* base, int ld, int row0, int k0) {
  int lane = threadIdx.x & 31;
  int r = row0 + (lane & 15);
  int hi = lane >> 4;
  const bf16_t* rp = base + r * ld + k0 + (hi << 3);
  bf16x8 lo = *(const bf16x8*)rp;
  bf16x8 hv = *(const bf16x8*)(rp + 16);
  return __builtin_shufflevector(lo, hv, 0, 1, 2, 3, 4, 5, 6, 7,
                                 8, 9, 10, 11, 12, 13, 14, 15);
}

// ---------------------------------------------------------------------------
// Generic batched bf16 WMMA GEMM:  C[z] = act( scale * A[z] @ B[z]^T + bias )
// A is [M,K] (lda), B is stored K-major as [N,K] (ldb) -- weights are
// pre-transposed on conversion. Tile 64(M) x 128(N) x 32(K), 8 waves (2x4),
// wave tile 32x32 via 4 WMMA. offX(z) = (z/xCnt)*xOuter + (z%xCnt)*xInner.
// ---------------------------------------------------------------------------
#define BMg 64
#define BNg 128
#define BKg 32

__global__ __launch_bounds__(256) void gemm_bf16_kernel(
    const bf16_t* __restrict__ A, int lda, long aO, long aI, int aC,
    const bf16_t* __restrict__ Bm, int ldb, long bO, long bI, int bC,
    float* __restrict__ C, int ldc, long cO, long cI, int cC,
    bf16_t* __restrict__ Cbf,
    const float* __restrict__ bias, int act, float scale,
    int M, int N, int Kd)
{
  __shared__ __align__(16) bf16_t As[BMg][LDK];
  __shared__ __align__(16) bf16_t Bt[BNg][LDK];
  int t = threadIdx.x;
  int z = blockIdx.z;
  long offA = (long)(z / aC) * aO + (long)(z % aC) * aI;
  long offB = (long)(z / bC) * bO + (long)(z % bC) * bI;
  long offC = (long)(z / cC) * cO + (long)(z % cC) * cI;
  const bf16_t* Ab = A + offA;
  const bf16_t* Bb = Bm + offB;
  int mBase = blockIdx.y * BMg;
  int nBase = blockIdx.x * BNg;
  int wid = t >> 5, lane = t & 31;
  int wm = wid >> 2, wn = wid & 3;          // 2 x 4 wave grid
  v8f acc[2][2] = {};

  for (int k0 = 0; k0 < Kd; k0 += BKg) {
    __syncthreads();
    { // stage A tile (64x32): one bf16x8 per thread
      int r = t >> 2, kb = (t & 3) * 8;
      int gr = mBase + r;
      if (gr < M) {
        const bf16_t* p = Ab + (long)gr * lda + k0 + kb;
        *(bf16x8*)&As[r][kb] = *(const bf16x8*)p;
        if (k0 + BKg < Kd) __builtin_prefetch(p + BKg, 0, 1);   // global_prefetch
      } else {
#pragma unroll
        for (int i = 0; i < 8; ++i) As[r][kb + i] = (bf16_t)0.f;
      }
    }
    { // stage B tile (128 n x 32 k) from [N,K] memory: two bf16x8 per thread
      int n = t >> 1, kb = (t & 1) * 16;
      int gc = nBase + n;
      if (gc < N) {
        const bf16_t* p = Bb + (long)gc * ldb + k0 + kb;
        *(bf16x8*)&Bt[n][kb]     = *(const bf16x8*)p;
        *(bf16x8*)&Bt[n][kb + 8] = *(const bf16x8*)(p + 8);
      } else {
#pragma unroll
        for (int i = 0; i < 16; ++i) Bt[n][kb + i] = (bf16_t)0.f;
      }
    }
    __syncthreads();

    v16bf a0 = frag_row(&As[0][0], LDK, wm * 32, 0);
    v16bf a1 = frag_row(&As[0][0], LDK, wm * 32 + 16, 0);
    v16bf b0 = frag_row(&Bt[0][0], LDK, wn * 32, 0);
    v16bf b1 = frag_row(&Bt[0][0], LDK, wn * 32 + 16, 0);
    acc[0][0] = WMMA_BF16(a0, b0, acc[0][0]);
    acc[0][1] = WMMA_BF16(a0, b1, acc[0][1]);
    acc[1][0] = WMMA_BF16(a1, b0, acc[1][0]);
    acc[1][1] = WMMA_BF16(a1, b1, acc[1][1]);
  }

  float* Cp = C + offC;
  bf16_t* Cq = Cbf ? Cbf + offC : (bf16_t*)0;
#pragma unroll
  for (int i = 0; i < 2; ++i)
#pragma unroll
    for (int j = 0; j < 2; ++j)
#pragma unroll
      for (int v = 0; v < 8; ++v) {
        int r = mBase + wm * 32 + i * 16 + ((lane >> 4) << 3) + v;
        int c = nBase + wn * 32 + j * 16 + (lane & 15);
        if (r < M && c < N) {
          float val = acc[i][j][v] * scale;
          if (bias) val += bias[c];
          if (act == 1) val = gelu_f(val);
          else if (act == 2) val = sigmoid_f(val);
          Cp[(long)r * ldc + c] = val;
          if (Cq) Cq[(long)r * ldc + c] = (bf16_t)val;
        }
      }
}

// ---------------------------------------------------------------------------
// Fused MoE expert FFN: one block = (32-token tile, expert). h kept in LDS.
//   h = gelu(X @ W1 + b1) ; routed += gate * (h @ W2 + b2)   (global f32 atomics)
// Expert weights are pre-transposed to [N,K]. Experts 0..31 routed (gate from
// wfull, early-exit when tile routes nothing), 32..39 shared (gate = 1).
// ---------------------------------------------------------------------------
__global__ __launch_bounds__(256) void moe_ffn_kernel(
    const bf16_t* __restrict__ X,      // [4096,512] bf16 (q2)
    const float* __restrict__ wfull,   // [4096,32]
    const bf16_t* __restrict__ EW1, const float* __restrict__ EB1,
    const bf16_t* __restrict__ EW2, const float* __restrict__ EB2,
    const bf16_t* __restrict__ SW1, const float* __restrict__ SB1,
    const bf16_t* __restrict__ SW2, const float* __restrict__ SB2,
    float* __restrict__ routed)        // [4096,512] f32 accum
{
  __shared__ __align__(16) bf16_t Xs[32][LDK];
  __shared__ __align__(16) bf16_t Wt[128][LDK];
  __shared__ __align__(16) bf16_t Hs[32][520];   // 520*2 B = 65*16 -> aligned rows
  __shared__ float gates[32];
  __shared__ int anyFlag;

  int t = threadIdx.x;
  int e = blockIdx.y;
  int t0 = blockIdx.x * 32;
  bool shd = (e >= NEXP);

  if (t == 0) anyFlag = 0;
  __syncthreads();
  if (t < 32) {
    float g = shd ? 1.f : wfull[(long)(t0 + t) * NEXP + e];
    gates[t] = g;
    if (g != 0.f) atomicOr(&anyFlag, 1);
  }
  __syncthreads();
  if (!anyFlag) return;   // tile routes nothing to this expert

  const bf16_t* W1 = shd ? SW1 + (long)(e - NEXP) * 262144 : EW1 + (long)e * 262144;
  const bf16_t* W2 = shd ? SW2 + (long)(e - NEXP) * 262144 : EW2 + (long)e * 262144;
  const float*  b1 = shd ? SB1 + (e - NEXP) * DMODEL : EB1 + e * DMODEL;
  const float*  b2 = shd ? SB2 + (e - NEXP) * DMODEL : EB2 + e * DMODEL;

  int wid = t >> 5, lane = t & 31;
  int wm = wid >> 2, wn = wid & 3;   // 2 x 4 waves, wave tile 16 x 32

  // -------- phase 1: h = gelu(X @ W1 + b1) -> Hs (bf16 in LDS) --------
  for (int n0 = 0; n0 < DMODEL; n0 += 128) {
    v8f acc0 = {}, acc1 = {};
    for (int k0 = 0; k0 < DMODEL; k0 += BKg) {
      __syncthreads();
      { int r = t >> 3, kb = (t & 7) * 4;
        *(bf16x4*)&Xs[r][kb] = *(const bf16x4*)(X + (long)(t0 + r) * DMODEL + k0 + kb); }
      { int n = t >> 1, kb = (t & 1) * 16;
        const bf16_t* p = W1 + (long)(n0 + n) * DMODEL + k0 + kb;
        *(bf16x8*)&Wt[n][kb]     = *(const bf16x8*)p;
        *(bf16x8*)&Wt[n][kb + 8] = *(const bf16x8*)(p + 8); }
      __syncthreads();
      v16bf a   = frag_row(&Xs[0][0], LDK, wm * 16, 0);
      v16bf bb0 = frag_row(&Wt[0][0], LDK, wn * 32, 0);
      v16bf bb1 = frag_row(&Wt[0][0], LDK, wn * 32 + 16, 0);
      acc0 = WMMA_BF16(a, bb0, acc0);
      acc1 = WMMA_BF16(a, bb1, acc1);
    }
    __syncthreads();
#pragma unroll
    for (int j = 0; j < 2; ++j) {
      v8f* ac = j ? &acc1 : &acc0;
#pragma unroll
      for (int v = 0; v < 8; ++v) {
        int r = wm * 16 + ((lane >> 4) << 3) + v;
        int c = n0 + wn * 32 + j * 16 + (lane & 15);
        Hs[r][c] = (bf16_t)gelu_f((*ac)[v] + b1[c]);
      }
    }
  }
  __syncthreads();

  // -------- phase 2: routed += gate * (Hs @ W2 + b2) --------
  for (int n0 = 0; n0 < DMODEL; n0 += 128) {
    v8f acc0 = {}, acc1 = {};
    for (int k0 = 0; k0 < DMODEL; k0 += BKg) {
      __syncthreads();
      { int n = t >> 1, kb = (t & 1) * 16;
        const bf16_t* p = W2 + (long)(n0 + n) * DMODEL + k0 + kb;
        *(bf16x8*)&Wt[n][kb]     = *(const bf16x8*)p;
        *(bf16x8*)&Wt[n][kb + 8] = *(const bf16x8*)(p + 8); }
      __syncthreads();
      v16bf a   = frag_row(&Hs[0][0], 520, wm * 16, k0);
      v16bf bb0 = frag_row(&Wt[0][0], LDK, wn * 32, 0);
      v16bf bb1 = frag_row(&Wt[0][0], LDK, wn * 32 + 16, 0);
      acc0 = WMMA_BF16(a, bb0, acc0);
      acc1 = WMMA_BF16(a, bb1, acc1);
    }
    __syncthreads();
#pragma unroll
    for (int j = 0; j < 2; ++j) {
      v8f* ac = j ? &acc1 : &acc0;
#pragma unroll
      for (int v = 0; v < 8; ++v) {
        int r = wm * 16 + ((lane >> 4) << 3) + v;
        int c = n0 + wn * 32 + j * 16 + (lane & 15);
        float g = gates[r];
        if (g != 0.f) {
          float val = g * ((*ac)[v] + b2[c]);
          atomicAdd(&routed[(long)(t0 + r) * DMODEL + c], val);
        }
      }
    }
  }
}

// ---------------------------------------------------------------------------
// LayerNorm over 512 cols: y = LN((X [+Res]) [*Mul_row/div]) * g + b
// ---------------------------------------------------------------------------
__global__ __launch_bounds__(256) void ln_kernel(
    const float* __restrict__ X, const float* __restrict__ Res,
    const float* __restrict__ Mul, int mulDiv,
    const float* __restrict__ g, const float* __restrict__ bv,
    float* __restrict__ Y, bf16_t* __restrict__ Ybf)
{
  long row = blockIdx.x;
  int t = threadIdx.x;
  const float* xr = X + row * DMODEL;
  float x0 = xr[t], x1 = xr[t + 256];
  if (Res) { x0 += Res[row * DMODEL + t]; x1 += Res[row * DMODEL + t + 256]; }
  if (Mul) { const float* m = Mul + (row / mulDiv) * DMODEL; x0 *= m[t]; x1 *= m[t + 256]; }
  __shared__ float red[256];
  red[t] = x0 + x1;
  __syncthreads();
  for (int s = 128; s > 0; s >>= 1) { if (t < s) red[t] += red[t + s]; __syncthreads(); }
  float mean = red[0] * (1.f / DMODEL);
  __syncthreads();
  float d0 = x0 - mean, d1 = x1 - mean;
  red[t] = d0 * d0 + d1 * d1;
  __syncthreads();
  for (int s = 128; s > 0; s >>= 1) { if (t < s) red[t] += red[t + s]; __syncthreads(); }
  float inv = rsqrtf(red[0] * (1.f / DMODEL) + 1e-5f);
  float y0 = d0 * inv * g[t] + bv[t];
  float y1 = d1 * inv * g[t + 256] + bv[t + 256];
  if (Y)   { Y[row * DMODEL + t] = y0; Y[row * DMODEL + t + 256] = y1; }
  if (Ybf) { Ybf[row * DMODEL + t] = (bf16_t)y0; Ybf[row * DMODEL + t + 256] = (bf16_t)y1; }
}

// Row softmax over 1024 (attention), writes bf16 probs.
__global__ __launch_bounds__(256) void softmax_kernel(
    const float* __restrict__ S, bf16_t* __restrict__ P)
{
  long row = blockIdx.x;
  int t = threadIdx.x;
  const float* sr = S + row * 1024;
  float v0 = sr[t], v1 = sr[t + 256], v2 = sr[t + 512], v3 = sr[t + 768];
  __shared__ float red[256];
  red[t] = fmaxf(fmaxf(v0, v1), fmaxf(v2, v3));
  __syncthreads();
  for (int s = 128; s > 0; s >>= 1) { if (t < s) red[t] = fmaxf(red[t], red[t + s]); __syncthreads(); }
  float m = red[0];
  __syncthreads();
  float e0 = expf(v0 - m), e1 = expf(v1 - m), e2 = expf(v2 - m), e3 = expf(v3 - m);
  red[t] = e0 + e1 + e2 + e3;
  __syncthreads();
  for (int s = 128; s > 0; s >>= 1) { if (t < s) red[t] += red[t + s]; __syncthreads(); }
  float inv = 1.f / red[0];
  bf16_t* pr = P + row * 1024;
  pr[t] = (bf16_t)(e0 * inv); pr[t + 256] = (bf16_t)(e1 * inv);
  pr[t + 512] = (bf16_t)(e2 * inv); pr[t + 768] = (bf16_t)(e3 * inv);
}

// Router: per-token softmax over 32 experts (1 wave = 1 token, lane = expert),
// top-4 via iterative wave argmax; accumulate aux stats with atomics.
__global__ __launch_bounds__(128) void router_topk_kernel(
    const float* __restrict__ logits, float* __restrict__ wfull,
    float* __restrict__ cnt, float* __restrict__ sump)
{
  int t = threadIdx.x, lane = t & 31;
  long tok = (long)blockIdx.x * 4 + (t >> 5);
  float l = logits[tok * NEXP + lane];
  float m = l;
  for (int o = 16; o > 0; o >>= 1) m = fmaxf(m, __shfl_xor(m, o, 32));
  float p = expf(l - m);
  float s = p;
  for (int o = 16; o > 0; o >>= 1) s += __shfl_xor(s, o, 32);
  p = p / s;
  bool taken = false;
  float topsum = 0.f;
  for (int it = 0; it < TOPK; ++it) {
    float cand = taken ? -1.f : p;
    float cm = cand;
    for (int o = 16; o > 0; o >>= 1) cm = fmaxf(cm, __shfl_xor(cm, o, 32));
    unsigned long long bal = __ballot(cand == cm && !taken);
    int winner = __ffsll((long long)bal) - 1;
    if (lane == winner) taken = true;
    topsum += cm;
  }
  float w = taken ? p / topsum : 0.f;
  wfull[tok * NEXP + lane] = w;
  atomicAdd(&cnt[lane], taken ? 1.f : 0.f);
  atomicAdd(&sump[lane], p);
}

__global__ void aux_final_kernel(const float* cnt, const float* sump, float* out) {
  if (threadIdx.x == 0 && blockIdx.x == 0) {
    float a = 0.f;
    for (int e = 0; e < NEXP; ++e)
      a += (cnt[e] * ((float)NEXP / TOPK) / (float)NTOK) * (sump[e] / (float)NTOK);
    out[0] = a;
  }
}

// key_source [B,C=1024,32*32] -> key_flat bf16 [(b*1024+pix), c]
__global__ void key_tr_kernel(const float* __restrict__ ks, bf16_t* __restrict__ kf) {
  long i = (long)blockIdx.x * 256 + threadIdx.x;
  if (i >= (long)BATCH * 1024 * 1024) return;
  long c = i & 1023, pix = (i >> 10) & 1023, b = i >> 20;
  kf[i] = (bf16_t)ks[(b * 1024 + c) * 1024 + pix];
}

// value_source [B,512,16,16] -> bilinear x2 (half-pixel) -> v_flat bf16 [(b*1024+pix), c]
__global__ void bilinear_kernel(const float* __restrict__ vs, bf16_t* __restrict__ vf) {
  long i = (long)blockIdx.x * 256 + threadIdx.x;
  if (i >= (long)BATCH * 1024 * VIN_) return;
  int c = (int)(i & 511);
  int pix = (int)((i >> 9) & 1023);
  int b = (int)(i >> 19);
  int oy = pix >> 5, ox = pix & 31;
  float fy = oy * 0.5f - 0.25f, fx = ox * 0.5f - 0.25f;
  int y0 = (int)floorf(fy), x0 = (int)floorf(fx);
  float wy = fy - y0, wx = fx - x0;
  int y0c = min(max(y0, 0), 15), y1c = min(max(y0 + 1, 0), 15);
  int x0c = min(max(x0, 0), 15), x1c = min(max(x0 + 1, 0), 15);
  const float* base = vs + ((long)b * VIN_ + c) * 256;
  float v00 = base[y0c * 16 + x0c], v01 = base[y0c * 16 + x1c];
  float v10 = base[y1c * 16 + x0c], v11 = base[y1c * 16 + x1c];
  float val = v00 * (1.f - wy) * (1.f - wx) + v01 * (1.f - wy) * wx +
              v10 * wy * (1.f - wx) + v11 * wy * wx;
  vf[i] = (bf16_t)val;
}

// plain f32 -> bf16
__global__ void cvt_kernel(const float* __restrict__ s, bf16_t* __restrict__ d, long n) {
  long i = (long)blockIdx.x * 256 + threadIdx.x;
  if (i < n) d[i] = (bf16_t)s[i];
}
// batched f32 [b,K,N] -> bf16 transposed [b,N,K]
__global__ void cvt_tr_kernel(const float* __restrict__ s, bf16_t* __restrict__ d,
                              int K, int N, long total) {
  long i = (long)blockIdx.x * 256 + threadIdx.x;
  if (i >= total) return;
  long kn = (long)K * N;
  long b = i / kn;
  long rem = i - b * kn;
  int k = (int)(rem % K);
  int n = (int)(rem / K);
  d[i] = (bf16_t)s[b * kn + (long)k * N + n];
}
// Pv f32 [tok(4096), 512] -> PvT bf16 [z=b*8+h][dh(64)][tok(1024)]
__global__ void pvt_kernel(const float* __restrict__ Pv, bf16_t* __restrict__ PvT) {
  long i = (long)blockIdx.x * 256 + threadIdx.x;
  if (i >= (long)BATCH * NHEAD * DHEAD * NQTOK) return;
  int tok = (int)(i & 1023);
  int dh  = (int)((i >> 10) & 63);
  int z   = (int)(i >> 16);
  int b = z >> 3, h = z & 7;
  PvT[i] = (bf16_t)Pv[((long)(b * NQTOK + tok)) * DMODEL + h * DHEAD + dh];
}
__global__ void zero_kernel(float* __restrict__ p, long n) {
  long i = (long)blockIdx.x * 256 + threadIdx.x;
  if (i < n) p[i] = 0.f;
}
__global__ void add_kernel(const float* __restrict__ a, const float* __restrict__ b,
                           float* __restrict__ o, long n) {
  long i = (long)blockIdx.x * 256 + threadIdx.x;
  if (i < n) o[i] = a[i] + b[i];
}
// mean over the 1024 token positions of out[b,n,d] -> bf16 [4,512]
__global__ void mean_kernel(const float* __restrict__ out, bf16_t* __restrict__ mb) {
  int i = blockIdx.x * 256 + threadIdx.x;
  if (i >= BATCH * DMODEL) return;
  int b = i >> 9, d = i & 511;
  float s = 0.f;
  for (int n = 0; n < NQTOK; ++n) s += out[((long)(b * NQTOK + n)) * DMODEL + d];
  mb[i] = (bf16_t)(s * (1.f / NQTOK));
}

// ---------------------------------------------------------------------------
// Host side
// ---------------------------------------------------------------------------
static inline void launch_gemm(hipStream_t st,
    const bf16_t* A, int lda, long aO, long aI, int aC,
    const bf16_t* B, int ldb, long bO, long bI, int bC,
    float* C, int ldc, long cO, long cI, int cC,
    bf16_t* Cbf, const float* bias, int act, float scale,
    int M, int N, int K, int nz)
{
  dim3 g((N + BNg - 1) / BNg, (M + BMg - 1) / BMg, nz);
  gemm_bf16_kernel<<<g, dim3(256), 0, st>>>(A, lda, aO, aI, aC, B, ldb, bO, bI, bC,
                                            C, ldc, cO, cI, cC, Cbf, bias, act, scale, M, N, K);
}
static inline void launch_cvt(hipStream_t st, const float* s, bf16_t* d, long n) {
  cvt_kernel<<<dim3((unsigned)((n + 255) / 256)), dim3(256), 0, st>>>(s, d, n);
}
static inline void launch_cvt_tr(hipStream_t st, const float* s, bf16_t* d,
                                 int K, int N, int batch) {
  long total = (long)batch * K * N;
  cvt_tr_kernel<<<dim3((unsigned)((total + 255) / 256)), dim3(256), 0, st>>>(s, d, K, N, total);
}

// d_in indices per setup_inputs() dict order
enum {
  I_GATEIN = 0, I_QUERY, I_KEYSRC, I_VALSRC,
  P_GATE_RED_W, P_GATE_RED_B, P_GATE_EMB_W, P_GATE_EMB_B,
  P_KEY_RED_W, P_KEY_RED_B, P_VAL_RED_W, P_VAL_RED_B,
  P_OG_W, P_OG_B,
  P_LNQ_G, P_LNQ_B, P_LNK_G, P_LNK_B, P_LNV_G, P_LNV_B,
  P_LN1_G, P_LN1_B, P_LN2_G, P_LN2_B,
  P_SAQ_W, P_SAQ_B, P_SAK_W, P_SAK_B, P_SAV_W, P_SAV_B, P_SAO_W, P_SAO_B,
  P_CAQ_W, P_CAQ_B, P_CAK_W, P_CAK_B, P_CAV_W, P_CAV_B, P_CAO_W, P_CAO_B,
  P_ROUTER_W1, P_ROUTER_W2,
  P_EXP_W1, P_EXP_B1, P_EXP_W2, P_EXP_B2,
  P_SH_W1, P_SH_B1, P_SH_W2, P_SH_B2
};

extern "C" void kernel_launch(void* const* d_in, const int* in_sizes, int n_in,
                              void* d_out, int out_size, void* d_ws, size_t ws_size,
                              hipStream_t stream) {
  (void)in_sizes; (void)n_in; (void)out_size; (void)ws_size;
  const float* F[64];
  for (int i = 0; i < 50; ++i) F[i] = (const float*)d_in[i];
  float* outp = (float*)d_out;

  // ---- workspace bump allocator ----
  char* wsb = (char*)d_ws;
  size_t off = 0;
  auto alloc = [&](size_t bytes) -> char* {
    char* p = wsb + off;
    off = (off + bytes + 255) & ~(size_t)255;
    return p;
  };
  auto abf = [&](long n) -> bf16_t* { return (bf16_t*)alloc((size_t)n * 2); };
  auto af32 = [&](long n) -> float* { return (float*)alloc((size_t)n * 4); };

  const long DD = (long)DMODEL * DMODEL;      // 262144
  // bf16 weights (all pre-transposed to [N,K])
  bf16_t* wGateRed = abf((long)GIN_ * GDIM);
  bf16_t* wGateEmb = abf((long)GDIM * DMODEL);
  bf16_t* wKeyRed  = abf((long)KIN_ * DMODEL);
  bf16_t* wValRed  = abf((long)VIN_ * DMODEL);
  bf16_t* wOg      = abf((long)DMODEL * GDIM);
  bf16_t* wSa[4], *wCa[4];
  for (int i = 0; i < 4; ++i) wSa[i] = abf(DD);
  for (int i = 0; i < 4; ++i) wCa[i] = abf(DD);
  bf16_t* wR1 = abf((long)DMODEL * RRANK);
  bf16_t* wR2 = abf((long)RRANK * NEXP);
  bf16_t* wE1 = abf((long)NEXP * DD);
  bf16_t* wE2 = abf((long)NEXP * DD);
  bf16_t* wS1 = abf((long)NSH * DD);
  bf16_t* wS2 = abf((long)NSH * DD);
  // bf16 activations
  bf16_t* gateInBf  = abf((long)BATCH * GIN_);
  bf16_t* gateLrBf  = abf((long)BATCH * GDIM);
  bf16_t* keyFlatBf = abf((long)NTOK * KIN_);
  bf16_t* vFlatBf   = abf((long)NTOK * VIN_);
  bf16_t* qnBf = abf((long)NTOK * DMODEL);
  bf16_t* kBf  = abf((long)NTOK * DMODEL);
  bf16_t* vBf  = abf((long)NTOK * DMODEL);
  bf16_t* PqBf = abf((long)NTOK * DMODEL);
  bf16_t* PkBf = abf((long)NTOK * DMODEL);
  bf16_t* PvT  = abf((long)BATCH * NHEAD * DHEAD * NQTOK);  // [z][dh][tok]
  bf16_t* probsBf = abf((long)BATCH * NHEAD * NQTOK * 1024);
  bf16_t* attnoBf = abf((long)NTOK * DMODEL);
  bf16_t* q1Bf = abf((long)NTOK * DMODEL);
  bf16_t* q2Bf = abf((long)NTOK * DMODEL);
  bf16_t* r1Bf = abf((long)NTOK * RRANK);
  bf16_t* meanBf = abf((long)BATCH * DMODEL);
  // f32 buffers
  float* gateLr  = af32((long)BATCH * GDIM);
  float* gateMod = af32((long)BATCH * DMODEL);
  float* kproj = af32((long)NTOK * DMODEL);
  float* vproj = af32((long)NTOK * DMODEL);
  float* Pq = af32((long)NTOK * DMODEL);
  float* Pk = af32((long)NTOK * DMODEL);
  float* Pv = af32((long)NTOK * DMODEL);
  float* scores = af32((long)BATCH * NHEAD * NQTOK * 1024);
  float* attno  = af32((long)NTOK * DMODEL);
  float* projo  = af32((long)NTOK * DMODEL);   // o-proj out (sa then ca)
  float* q1 = af32((long)NTOK * DMODEL);
  float* q2 = af32((long)NTOK * DMODEL);
  float* r1 = af32((long)NTOK * RRANK);
  float* logits = af32((long)NTOK * NEXP);
  float* wfull  = af32((long)NTOK * NEXP);
  float* routed = af32((long)NTOK * DMODEL);
  float* auxacc = af32(64);                    // [0..31]=cnt, [32..63]=sumP

  const float* query = F[I_QUERY];

  // ---- 0: weight conversion to bf16, transposed [K,N] -> [N,K] ----
  launch_cvt_tr(stream, F[P_GATE_RED_W], wGateRed, GIN_, GDIM, 1);
  launch_cvt_tr(stream, F[P_GATE_EMB_W], wGateEmb, GDIM, DMODEL, 1);
  launch_cvt_tr(stream, F[P_KEY_RED_W], wKeyRed, KIN_, DMODEL, 1);
  launch_cvt_tr(stream, F[P_VAL_RED_W], wValRed, VIN_, DMODEL, 1);
  launch_cvt_tr(stream, F[P_OG_W], wOg, DMODEL, GDIM, 1);
  launch_cvt_tr(stream, F[P_SAQ_W], wSa[0], DMODEL, DMODEL, 1);
  launch_cvt_tr(stream, F[P_SAK_W], wSa[1], DMODEL, DMODEL, 1);
  launch_cvt_tr(stream, F[P_SAV_W], wSa[2], DMODEL, DMODEL, 1);
  launch_cvt_tr(stream, F[P_SAO_W], wSa[3], DMODEL, DMODEL, 1);
  launch_cvt_tr(stream, F[P_CAQ_W], wCa[0], DMODEL, DMODEL, 1);
  launch_cvt_tr(stream, F[P_CAK_W], wCa[1], DMODEL, DMODEL, 1);
  launch_cvt_tr(stream, F[P_CAV_W], wCa[2], DMODEL, DMODEL, 1);
  launch_cvt_tr(stream, F[P_CAO_W], wCa[3], DMODEL, DMODEL, 1);
  launch_cvt_tr(stream, F[P_ROUTER_W1], wR1, DMODEL, RRANK, 1);
  launch_cvt_tr(stream, F[P_ROUTER_W2], wR2, RRANK, NEXP, 1);
  launch_cvt_tr(stream, F[P_EXP_W1], wE1, DMODEL, DMODEL, NEXP);
  launch_cvt_tr(stream, F[P_EXP_W2], wE2, DMODEL, DMODEL, NEXP);
  launch_cvt_tr(stream, F[P_SH_W1], wS1, DMODEL, DMODEL, NSH);
  launch_cvt_tr(stream, F[P_SH_W2], wS2, DMODEL, DMODEL, NSH);
  launch_cvt(stream, F[I_GATEIN], gateInBf, (long)BATCH * GIN_);

  // ---- 1: zero accumulators ----
  zero_kernel<<<dim3((NTOK * DMODEL + 255) / 256), dim3(256), 0, stream>>>(routed, (long)NTOK * DMODEL);
  zero_kernel<<<dim3(1), dim3(256), 0, stream>>>(auxacc, 64);

  // ---- 2: gate stream ----
  launch_gemm(stream, gateInBf, GIN_, 0, 0, 1, wGateRed, GIN_, 0, 0, 1,
              gateLr, GDIM, 0, 0, 1, gateLrBf, F[P_GATE_RED_B], 0, 1.f,
              BATCH, GDIM, GIN_, 1);
  launch_gemm(stream, gateLrBf, GDIM, 0, 0, 1, wGateEmb, GDIM, 0, 0, 1,
              gateMod, DMODEL, 0, 0, 1, (bf16_t*)0, F[P_GATE_EMB_B], 0, 1.f,
              BATCH, DMODEL, GDIM, 1);

  // ---- 3: key stream ----
  key_tr_kernel<<<dim3((unsigned)(((long)BATCH * 1024 * 1024 + 255) / 256)), dim3(256), 0, stream>>>(
      F[I_KEYSRC], keyFlatBf);
  launch_gemm(stream, keyFlatBf, KIN_, 0, 0, 1, wKeyRed, KIN_, 0, 0, 1,
              kproj, DMODEL, 0, 0, 1, (bf16_t*)0, F[P_KEY_RED_B], 0, 1.f,
              NTOK, DMODEL, KIN_, 1);
  ln_kernel<<<dim3(NTOK), dim3(256), 0, stream>>>(kproj, (const float*)0, (const float*)0, 1,
      F[P_LNK_G], F[P_LNK_B], (float*)0, kBf);

  // ---- 4: value stream (bilinear resize + reduce + gate-mod LN) ----
  bilinear_kernel<<<dim3((unsigned)(((long)BATCH * 1024 * VIN_ + 255) / 256)), dim3(256), 0, stream>>>(
      F[I_VALSRC], vFlatBf);
  launch_gemm(stream, vFlatBf, VIN_, 0, 0, 1, wValRed, VIN_, 0, 0, 1,
              vproj, DMODEL, 0, 0, 1, (bf16_t*)0, F[P_VAL_RED_B], 0, 1.f,
              NTOK, DMODEL, VIN_, 1);
  ln_kernel<<<dim3(NTOK), dim3(256), 0, stream>>>(vproj, (const float*)0, gateMod, NQTOK,
      F[P_LNV_G], F[P_LNV_B], (float*)0, vBf);

  // ---- 5: qn = LN(query) ----
  ln_kernel<<<dim3(NTOK), dim3(256), 0, stream>>>(query, (const float*)0, (const float*)0, 1,
      F[P_LNQ_G], F[P_LNQ_B], (float*)0, qnBf);

  // ---- attention helper (same shapes for SA and CA) ----
  const long hOut = (long)NQTOK * DMODEL;   // per-batch stride 524288
  auto attention = [&](const bf16_t* qin, const bf16_t* kin, const bf16_t* vin,
                       const bf16_t* wq, const float* bq, const bf16_t* wk, const float* bk,
                       const bf16_t* wv, const float* bvv, const bf16_t* wo, const float* bo,
                       float* oOut) {
    launch_gemm(stream, qin, DMODEL, 0, 0, 1, wq, DMODEL, 0, 0, 1,
                Pq, DMODEL, 0, 0, 1, PqBf, bq, 0, 1.f, NTOK, DMODEL, DMODEL, 1);
    launch_gemm(stream, kin, DMODEL, 0, 0, 1, wk, DMODEL, 0, 0, 1,
                Pk, DMODEL, 0, 0, 1, PkBf, bk, 0, 1.f, NTOK, DMODEL, DMODEL, 1);
    launch_gemm(stream, vin, DMODEL, 0, 0, 1, wv, DMODEL, 0, 0, 1,
                Pv, DMODEL, 0, 0, 1, (bf16_t*)0, bvv, 0, 1.f, NTOK, DMODEL, DMODEL, 1);
    pvt_kernel<<<dim3((unsigned)(((long)BATCH * NHEAD * DHEAD * NQTOK + 255) / 256)),
                 dim3(256), 0, stream>>>(Pv, PvT);
    // scores[z=b*8+h] = (1/8) * Qh @ Kh^T : B = PkBf rows are [token, 512] (K-major)
    launch_gemm(stream, PqBf, DMODEL, hOut, DHEAD, NHEAD,
                PkBf, DMODEL, hOut, DHEAD, NHEAD,
                scores, 1024, (long)NQTOK * 1024, 0, 1,
                (bf16_t*)0, (const float*)0, 0, 0.125f, NQTOK, 1024, DHEAD, BATCH * NHEAD);
    softmax_kernel<<<dim3(BATCH * NHEAD * NQTOK), dim3(256), 0, stream>>>(scores, probsBf);
    // attno[z] = P @ Vh : B = PvT [z][dh][tok] (K-major rows of length 1024)
    launch_gemm(stream, probsBf, 1024, (long)NQTOK * 1024, 0, 1,
                PvT, NQTOK, (long)DHEAD * NQTOK, 0, 1,
                attno, DMODEL, hOut, DHEAD, NHEAD,
                attnoBf, (const float*)0, 0, 1.f, NQTOK, DHEAD, 1024, BATCH * NHEAD);
    launch_gemm(stream, attnoBf, DMODEL, 0, 0, 1, wo, DMODEL, 0, 0, 1,
                oOut, DMODEL, 0, 0, 1, (bf16_t*)0, bo, 0, 1.f, NTOK, DMODEL, DMODEL, 1);
  };

  // ---- 6: self attention + q1 = LN(query + sa) ----
  attention(qnBf, qnBf, qnBf, wSa[0], F[P_SAQ_B], wSa[1], F[P_SAK_B],
            wSa[2], F[P_SAV_B], wSa[3], F[P_SAO_B], projo);
  ln_kernel<<<dim3(NTOK), dim3(256), 0, stream>>>(projo, query, (const float*)0, 1,
      F[P_LN1_G], F[P_LN1_B], q1, q1Bf);

  // ---- 7: cross attention + q2 = LN(q1 + ca) ----
  attention(q1Bf, kBf, vBf, wCa[0], F[P_CAQ_B], wCa[1], F[P_CAK_B],
            wCa[2], F[P_CAV_B], wCa[3], F[P_CAO_B], projo);
  ln_kernel<<<dim3(NTOK), dim3(256), 0, stream>>>(projo, q1, (const float*)0, 1,
      F[P_LN2_G], F[P_LN2_B], q2, q2Bf);

  // ---- 8: router (low-rank) + top-k gates + aux loss ----
  launch_gemm(stream, q2Bf, DMODEL, 0, 0, 1, wR1, DMODEL, 0, 0, 1,
              r1, RRANK, 0, 0, 1, r1Bf, (const float*)0, 0, 1.f, NTOK, RRANK, DMODEL, 1);
  launch_gemm(stream, r1Bf, RRANK, 0, 0, 1, wR2, RRANK, 0, 0, 1,
              logits, NEXP, 0, 0, 1, (bf16_t*)0, (const float*)0, 0, 1.f, NTOK, NEXP, RRANK, 1);
  router_topk_kernel<<<dim3(NTOK / 4), dim3(128), 0, stream>>>(logits, wfull, auxacc, auxacc + 32);
  aux_final_kernel<<<dim3(1), dim3(32), 0, stream>>>(auxacc, auxacc + 32, outp + AUX_OFF);

  // ---- 9: MoE (32 routed experts with early-exit + 8 shared experts) ----
  moe_ffn_kernel<<<dim3(NTOK / 32, NEXP + NSH), dim3(256), 0, stream>>>(
      q2Bf, wfull, wE1, F[P_EXP_B1], wE2, F[P_EXP_B2],
      wS1, F[P_SH_B1], wS2, F[P_SH_B2], routed);

  // ---- 10: out = q2 + moe; og = sigmoid(mean_n(out) @ og_w + og_b) ----
  add_kernel<<<dim3((NTOK * DMODEL + 255) / 256), dim3(256), 0, stream>>>(
      q2, routed, outp, (long)NTOK * DMODEL);
  mean_kernel<<<dim3((BATCH * DMODEL + 255) / 256), dim3(256), 0, stream>>>(outp, meanBf);
  launch_gemm(stream, meanBf, DMODEL, 0, 0, 1, wOg, DMODEL, 0, 0, 1,
              outp + OG_OFF, GDIM, 0, 0, 1, (bf16_t*)0, F[P_OG_B], 2, 1.f,
              BATCH, GDIM, DMODEL, 1);
}
